// SpikingGraphWaveletNet_38491496907292
// MI455X (gfx1250) — matrix operand order repaired and checked by reading, not compiled, not run on hardware
//
#include <hip/hip_runtime.h>

// ---------------- problem constants ----------------
constexpr int Bsz = 4;
constexpr int Npt = 1024;
constexpr int Kn  = 20;
constexpr int Tst = 10;
constexpr int Mrows = Bsz * Tst * Npt;   // 40960
constexpr int NClass = 40;

typedef __bf16 v16bf __attribute__((ext_vector_type(16)));
typedef float  v8f   __attribute__((ext_vector_type(8)));

union FragBF { v16bf v; unsigned u[8]; };

// ---------------- graph build: kNN + adaptive weights ----------------
__global__ void k_build_graph(const float* __restrict__ pc, int* __restrict__ idx,
                              float* __restrict__ w, float* __restrict__ s) {
  int gid = blockIdx.x * blockDim.x + threadIdx.x;
  if (gid >= Bsz * Npt) return;
  int b = gid / Npt, n = gid % Npt;
  const float* base = pc + (size_t)b * Npt * 3;
  float px = base[n * 3 + 0], py = base[n * 3 + 1], pz = base[n * 3 + 2];
  float dArr[Kn]; int iArr[Kn];
#pragma unroll
  for (int k = 0; k < Kn; ++k) { dArr[k] = 3.0e38f; iArr[k] = 0; }
  for (int j = 0; j < Npt; ++j) {
    float dx = base[j * 3 + 0] - px;
    float dy = base[j * 3 + 1] - py;
    float dz = base[j * 3 + 2] - pz;
    float d2 = dx * dx + dy * dy + dz * dz;
    if (j == n) d2 += 1.0e9f;
    if (d2 < dArr[Kn - 1]) {
      float dc = d2; int ic = j;
#pragma unroll
      for (int p = 0; p < Kn; ++p) {
        if (dc < dArr[p]) {
          float td = dArr[p]; int ti = iArr[p];
          dArr[p] = dc; iArr[p] = ic; dc = td; ic = ti;
        }
      }
    }
  }
  float msum = 0.f, d2c[Kn];
#pragma unroll
  for (int k = 0; k < Kn; ++k) {
    float dd = sqrtf(fmaxf(dArr[k], 1e-12f));
    d2c[k] = dd * dd; msum += dd;
  }
  float sigma = (msum / (float)Kn) + 1e-6f;     // BETA=1
  float inv_s2 = 1.0f / (sigma * sigma);
  float wsum = 0.f, wv[Kn];
#pragma unroll
  for (int k = 0; k < Kn; ++k) { wv[k] = __expf(-d2c[k] * inv_s2); wsum += wv[k]; }
  float inv = 1.0f / (wsum + 1e-6f);
#pragma unroll
  for (int k = 0; k < Kn; ++k) { idx[gid * Kn + k] = iArr[k]; w[gid * Kn + k] = wv[k] * inv; }
  s[gid] = sigma;                                // LAM=1
}

// ---------------- threefry2x32 (JAX default PRNG, key=(0,42)) ----------------
__device__ __forceinline__ unsigned rotl32(unsigned x, unsigned d) {
  return (x << d) | (x >> (32u - d));
}
__device__ __forceinline__ void threefry2x32(unsigned k0, unsigned k1,
                                             unsigned x0, unsigned x1,
                                             unsigned& o0, unsigned& o1) {
  unsigned k2 = k0 ^ k1 ^ 0x1BD11BDAu;
  x0 += k0; x1 += k1;
#define TF4(a,b,c,d) \
  x0 += x1; x1 = rotl32(x1,a); x1 ^= x0; \
  x0 += x1; x1 = rotl32(x1,b); x1 ^= x0; \
  x0 += x1; x1 = rotl32(x1,c); x1 ^= x0; \
  x0 += x1; x1 = rotl32(x1,d); x1 ^= x0;
  TF4(13,15,26,6)  x0 += k1; x1 += k2 + 1u;
  TF4(17,29,16,24) x0 += k2; x1 += k0 + 2u;
  TF4(13,15,26,6)  x0 += k0; x1 += k1 + 3u;
  TF4(17,29,16,24) x0 += k1; x1 += k2 + 4u;
  TF4(13,15,26,6)  x0 += k2; x1 += k0 + 5u;
#undef TF4
  o0 = x0; o1 = x1;
}

// ---------------- encoder + Poisson rate coding -> X[B,T,N,64] bf16 ----------------
__global__ void k_encode(const float* __restrict__ pc, const float* __restrict__ encW,
                         const float* __restrict__ encb, __bf16* __restrict__ X) {
  int gid = blockIdx.x * blockDim.x + threadIdx.x;
  if (gid >= Bsz * Npt * 64) return;
  int f = gid & 63;
  int n = (gid >> 6) % Npt;
  int b = gid / (Npt * 64);
  const float* p = pc + ((size_t)b * Npt + n) * 3;
  float acc = encb[f] + p[0] * encW[0 * 64 + f] + p[1] * encW[1 * 64 + f] + p[2] * encW[2 * 64 + f];
  acc = fmaxf(acc, 0.f);                  // relu
  float rate = fminf(acc, 1.f);           // clip to [0,1]
  const unsigned ntot = (unsigned)Bsz * Tst * Npt * 64;
  const unsigned half = ntot / 2u;
  for (int t = 0; t < Tst; ++t) {
    unsigned i = ((unsigned)((b * Tst + t) * Npt + n) << 6) + (unsigned)f;
    unsigned j = (i < half) ? i : (i - half);
    unsigned y0, y1;
    threefry2x32(0u, 42u, j, j + half, y0, y1);
    unsigned bits = (i < half) ? y0 : y1;
    float u = __uint_as_float((bits >> 9) | 0x3f800000u) - 1.0f;
    X[((size_t)(b * Tst + t) * Npt + n) * 64 + f] = (__bf16)((u < rate) ? 1.0f : 0.0f);
  }
}

// ---------------- Laplacian applies: stage A = [x | t1 | t2] bf16 ----------------
template <int F>
__global__ void k_lap1(const __bf16* __restrict__ X, const int* __restrict__ idx,
                       const float* __restrict__ w, const float* __restrict__ s,
                       __bf16* __restrict__ A) {
  int m = blockIdx.x;           // (b*T+t)*N + n
  int f = threadIdx.x;
  int b = m / (Tst * Npt);
  int n = m % Npt;
  __shared__ int   sI[Kn];
  __shared__ float sW[Kn];
  __shared__ float sS;
  if (f < Kn) { sI[f] = idx[(b * Npt + n) * Kn + f]; sW[f] = w[(b * Npt + n) * Kn + f]; }
  if (f == 0) sS = s[b * Npt + n];
  __syncthreads();
  const __bf16* Xt = X + (size_t)(m - n) * F;       // (b,t)-slab base
  float xv = (float)Xt[(size_t)n * F + f];
  float acc = 0.f;
#pragma unroll
  for (int k = 0; k < Kn; ++k) acc += sW[k] * (float)Xt[(size_t)sI[k] * F + f];
  float t1 = sS * (xv - acc);
  __bf16* Ar = A + (size_t)m * (3 * F);
  Ar[f]     = (__bf16)xv;
  Ar[F + f] = (__bf16)t1;
}

template <int F>
__global__ void k_lap2(const int* __restrict__ idx, const float* __restrict__ w,
                       const float* __restrict__ s, __bf16* __restrict__ A) {
  int m = blockIdx.x;
  int f = threadIdx.x;
  int b = m / (Tst * Npt);
  int n = m % Npt;
  __shared__ int   sI[Kn];
  __shared__ float sW[Kn];
  __shared__ float sS;
  if (f < Kn) { sI[f] = idx[(b * Npt + n) * Kn + f]; sW[f] = w[(b * Npt + n) * Kn + f]; }
  if (f == 0) sS = s[b * Npt + n];
  __syncthreads();
  const __bf16* Aslab = A + (size_t)(m - n) * (3 * F);
  float t1 = (float)A[(size_t)m * (3 * F) + F + f];
  float acc = 0.f;
#pragma unroll
  for (int k = 0; k < Kn; ++k) acc += sW[k] * (float)Aslab[(size_t)sI[k] * (3 * F) + F + f];
  float xv = (float)A[(size_t)m * (3 * F) + f];
  float t2 = 2.0f * sS * (t1 - acc) - xv;
  A[(size_t)m * (3 * F) + 2 * F + f] = (__bf16)t2;
}

// ---------------- W[3,F,G] f32 -> Wt[G,3F] bf16 ----------------
__global__ void k_convw(const float* __restrict__ W, __bf16* __restrict__ Wt, int F, int G) {
  int K3 = 3 * F;
  int gid = blockIdx.x * blockDim.x + threadIdx.x;
  if (gid >= G * K3) return;
  int k = gid % K3, g = gid / K3;
  Wt[(size_t)g * K3 + k] = (__bf16)W[(size_t)k * G + g];
}

// ---------------- WMMA bf16 GEMM: Y[M,G] = A[M,K3] * Wt^T + bias ----------------
// One wave computes a 32(M) x 64(N) strip: 2 A-frags x 4 B-frags -> 8 WMMAs per K-step
// against 12 b128 loads (compute:load doubled vs 16x64 blocking).
template <int K3, int G>
__global__ void __launch_bounds__(128) k_wmma_gemm(const __bf16* __restrict__ A,
                                                   const __bf16* __restrict__ Wt,
                                                   const float* __restrict__ bias,
                                                   float* __restrict__ Y) {
  int lane = threadIdx.x & 31;
  int wave = threadIdx.x >> 5;
  int tileM = blockIdx.y * 128 + wave * 32;
  int tileN = blockIdx.x * 64;
  int col  = lane & 15;
  int half = lane >> 4;

  const __bf16* Arow0 = A + (size_t)(tileM + col) * K3;
  const __bf16* Arow1 = A + (size_t)(tileM + 16 + col) * K3;

  v8f acc[2][4];
#pragma unroll
  for (int i = 0; i < 2; ++i)
#pragma unroll
    for (int j = 0; j < 4; ++j)
      acc[i][j] = (v8f){0.f,0.f,0.f,0.f,0.f,0.f,0.f,0.f};

  for (int kk = 0; kk < K3; kk += 32) {
    if (kk + 32 < K3) {
      __builtin_prefetch(Arow0 + kk + 32, 0, 3);
      __builtin_prefetch(Arow1 + kk + 32, 0, 3);
    }
    // A fragments, 16x32 bf16, ISA layout: V0..3 -> K{2v,2v+1}+8*half, V4..7 -> 16+...
    FragBF a[2];
#pragma unroll
    for (int v = 0; v < 8; ++v) {
      int base = ((v < 4) ? (v * 2) : (16 + (v - 4) * 2)) + half * 8;
      a[0].u[v] = *(const unsigned*)(Arow0 + kk + base);
      a[1].u[v] = *(const unsigned*)(Arow1 + kk + base);
    }
    // B fragments, 32x16 bf16: V0..3 -> K{2v,2v+1}+8*half, V4..7 -> 16+..., lane col = N
    FragBF bf[4];
#pragma unroll
    for (int j = 0; j < 4; ++j) {
      const __bf16* Brow = Wt + (size_t)(tileN + 16 * j + col) * K3 + kk;
#pragma unroll
      for (int v = 0; v < 8; ++v) {
        int base = ((v < 4) ? (v * 2) : (16 + (v - 4) * 2)) + half * 8;
        bf[j].u[v] = *(const unsigned*)(Brow + base);
      }
    }
#pragma unroll
    for (int j = 0; j < 4; ++j) {
#pragma unroll
      for (int i = 0; i < 2; ++i) {
        acc[i][j] = __builtin_amdgcn_wmma_f32_16x16x32_bf16(
            false, a[i].v, false, bf[j].v, (short)0, acc[i][j], false, false);
      }
    }
  }
  // epilogue: + bias, store f32. C/D layout: lane N=col, row = r + 8*half
#pragma unroll
  for (int i = 0; i < 2; ++i) {
#pragma unroll
    for (int j = 0; j < 4; ++j) {
      int g = tileN + 16 * j + col;
      float bv = bias[g];
#pragma unroll
      for (int r = 0; r < 8; ++r) {
        Y[(size_t)(tileM + 16 * i + r + 8 * half) * G + g] = acc[i][j][r] + bv;
      }
    }
  }
}

// ---------------- bipolar LIF scan over T -> X[B,T,N,2G] bf16 spikes ----------------
template <int G>
__global__ void k_lif(const float* __restrict__ Y, __bf16* __restrict__ X) {
  int gid = blockIdx.x * blockDim.x + threadIdx.x;
  if (gid >= Bsz * Npt * G) return;
  int g = gid % G;
  int n = (gid / G) % Npt;
  int b = gid / (Npt * G);
  const float decay = 0.9512294245007140f;   // exp(-1/20)
  float v = 0.f;
  for (int t = 0; t < Tst; ++t) {
    size_t o = (size_t)(b * Tst + t) * Npt + n;
    v = decay * v + Y[o * G + g];
    float sp = (v - 1.0f >= 0.f) ? 1.f : 0.f;   // spike(v - TH_POS)
    float sn = (-1.0f - v >= 0.f) ? 1.f : 0.f;  // spike(TH_NEG - v)
    v = v - sp + sn;                            // v - sp*TH_POS - sn*TH_NEG
    X[o * (2 * G) + g]     = (__bf16)sp;
    X[o * (2 * G) + G + g] = (__bf16)sn;
  }
}

// ---------------- mean over (t,n): pooled[B,512] ----------------
__global__ void k_pool(const __bf16* __restrict__ X, float* __restrict__ pooled) {
  int bf = blockIdx.x;               // b*512 + f
  int b = bf >> 9, f = bf & 511;
  float acc = 0.f;
  for (int i = threadIdx.x; i < Tst * Npt; i += blockDim.x) {
    acc += (float)X[((size_t)b * Tst * Npt + i) * 512 + f];
  }
  __shared__ float red[256];
  red[threadIdx.x] = acc;
  __syncthreads();
  for (int st = 128; st > 0; st >>= 1) {
    if (threadIdx.x < st) red[threadIdx.x] += red[threadIdx.x + st];
    __syncthreads();
  }
  if (threadIdx.x == 0) pooled[bf] = red[0] / (float)(Tst * Npt);
}

// ---------------- readout: out[B,40] = pooled @ ro_W + ro_b ----------------
__global__ void k_readout(const float* __restrict__ pooled, const float* __restrict__ roW,
                          const float* __restrict__ rob, float* __restrict__ out) {
  int gid = blockIdx.x * blockDim.x + threadIdx.x;
  if (gid >= Bsz * NClass) return;
  int c = gid % NClass, b = gid / NClass;
  float acc = rob[c];
  for (int f = 0; f < 512; ++f) acc += pooled[b * 512 + f] * roW[f * NClass + c];
  out[gid] = acc;
}

// ---------------- host orchestration ----------------
extern "C" void kernel_launch(void* const* d_in, const int* in_sizes, int n_in,
                              void* d_out, int out_size, void* d_ws, size_t ws_size,
                              hipStream_t stream) {
  (void)in_sizes; (void)n_in; (void)out_size; (void)ws_size;
  const float* pc   = (const float*)d_in[0];
  const float* encW = (const float*)d_in[1];
  const float* encb = (const float*)d_in[2];
  const float* W0   = (const float*)d_in[3];
  const float* b0   = (const float*)d_in[4];
  const float* W1   = (const float*)d_in[5];
  const float* b1   = (const float*)d_in[6];
  const float* W2   = (const float*)d_in[7];
  const float* b2   = (const float*)d_in[8];
  const float* roW  = (const float*)d_in[9];
  const float* rob  = (const float*)d_in[10];
  float* out = (float*)d_out;

  char* ws = (char*)d_ws;
  size_t off = 0;
  auto take = [&](size_t bytes) -> char* {
    char* p = ws + off;
    off = (off + bytes + 255) & ~(size_t)255;
    return p;
  };
  int*    idx    = (int*)   take((size_t)Bsz * Npt * Kn * 4);
  float*  wgt    = (float*) take((size_t)Bsz * Npt * Kn * 4);
  float*  sca    = (float*) take((size_t)Bsz * Npt * 4);
  __bf16* X      = (__bf16*)take((size_t)Mrows * 512 * 2);
  __bf16* A      = (__bf16*)take((size_t)Mrows * 768 * 2);
  float*  Y      = (float*) take((size_t)Mrows * 256 * 4);
  __bf16* Wt     = (__bf16*)take((size_t)256 * 768 * 2);
  float*  pooled = (float*) take((size_t)Bsz * 512 * 4);

  // graph + encoding
  k_build_graph<<<dim3((Bsz * Npt + 63) / 64), dim3(64), 0, stream>>>(pc, idx, wgt, sca);
  k_encode<<<dim3((Bsz * Npt * 64 + 255) / 256), dim3(256), 0, stream>>>(pc, encW, encb, X);

  // layer 0: F=64 -> G=64, K3=192
  k_convw<<<dim3((64 * 192 + 255) / 256), dim3(256), 0, stream>>>(W0, Wt, 64, 64);
  k_lap1<64><<<dim3(Mrows), dim3(64), 0, stream>>>(X, idx, wgt, sca, A);
  k_lap2<64><<<dim3(Mrows), dim3(64), 0, stream>>>(idx, wgt, sca, A);
  k_wmma_gemm<192, 64><<<dim3(1, Mrows / 128), dim3(128), 0, stream>>>(A, Wt, b0, Y);
  k_lif<64><<<dim3((Bsz * Npt * 64 + 255) / 256), dim3(256), 0, stream>>>(Y, X);

  // layer 1: F=128 -> G=128, K3=384
  k_convw<<<dim3((128 * 384 + 255) / 256), dim3(256), 0, stream>>>(W1, Wt, 128, 128);
  k_lap1<128><<<dim3(Mrows), dim3(128), 0, stream>>>(X, idx, wgt, sca, A);
  k_lap2<128><<<dim3(Mrows), dim3(128), 0, stream>>>(idx, wgt, sca, A);
  k_wmma_gemm<384, 128><<<dim3(2, Mrows / 128), dim3(128), 0, stream>>>(A, Wt, b1, Y);
  k_lif<128><<<dim3((Bsz * Npt * 128 + 255) / 256), dim3(256), 0, stream>>>(Y, X);

  // layer 2: F=256 -> G=256, K3=768
  k_convw<<<dim3((256 * 768 + 255) / 256), dim3(256), 0, stream>>>(W2, Wt, 256, 256);
  k_lap1<256><<<dim3(Mrows), dim3(256), 0, stream>>>(X, idx, wgt, sca, A);
  k_lap2<256><<<dim3(Mrows), dim3(256), 0, stream>>>(idx, wgt, sca, A);
  k_wmma_gemm<768, 256><<<dim3(4, Mrows / 128), dim3(128), 0, stream>>>(A, Wt, b2, Y);
  k_lif<256><<<dim3((Bsz * Npt * 256 + 255) / 256), dim3(256), 0, stream>>>(Y, X);

  // readout
  k_pool<<<dim3(Bsz * 512), dim3(256), 0, stream>>>(X, pooled);
  k_readout<<<dim3(1), dim3(256), 0, stream>>>(pooled, roW, rob, out);
}
